// ManualLSTM_46084999086429
// MI455X (gfx1250) — matrix operand order repaired and checked by reading, not compiled, not run on hardware
//
#include <hip/hip_runtime.h>
#include <stdint.h>

// ---------------------------------------------------------------------------
// LSTM forward S=512,B=64,I=H=512 on gfx1250 (wave32, WMMA bf16, TDM).
//  P0a: pack W/U fp32 -> bf16 column panels [ncat/16][16][K] (B-frag friendly)
//  P0b: convert x fp32 -> bf16 row-major (one-time)
//  P1 : gx = x @ Wcat + bcat. Per block: 4 adjacent B panels (64KB, contiguous)
//       staged to LDS via ONE tensor_load_to_lds; each A fragment feeds 4 WMMAs.
//       K-loop unroll capped at 2 so B frags stream from LDS (no spills).
//  P2 : persistent single-workgroup recurrence: h in LDS (bf16), c in VGPRs.
//  P3 : zero c0 output (reference returns the initial zeros).
// ---------------------------------------------------------------------------

typedef __bf16 v16bf __attribute__((ext_vector_type(16)));
typedef __bf16 v8bf  __attribute__((ext_vector_type(8)));
typedef float  v8f   __attribute__((ext_vector_type(8)));
typedef unsigned int v4u __attribute__((ext_vector_type(4)));
typedef int          v4i __attribute__((ext_vector_type(4)));
typedef int          v8i __attribute__((ext_vector_type(8)));

__device__ __forceinline__ unsigned short f2bf(float f) {
    union { float f; unsigned u; } c; c.f = f;
    unsigned r = c.u + 0x7FFFu + ((c.u >> 16) & 1u);   // round-to-nearest-even
    return (unsigned short)(r >> 16);
}
__device__ __forceinline__ float bf2f(unsigned u16) {
    union { unsigned u; float f; } c; c.u = u16 << 16;
    return c.f;
}
__device__ __forceinline__ float sigmoidf_(float x) {
    return 1.0f / (1.0f + __expf(-x));
}

static constexpr int S = 512, B = 64, I = 512, H = 512;
static constexpr int NCAT = 4 * H;          // 2048 gate columns
static constexpr int MTOT = S * B;          // 32768 rows
static constexpr int TN   = NCAT / 16;      // 128 column tiles
static constexpr int TM   = MTOT / 16;      // 2048 row tiles

// ---------------------------------------------------------------------------
// TDM: 1-D copy of nelems bf16 elements, global -> LDS (contiguous).
// D# group0: count=1 | lds_addr | global_addr | type=2.
// D# group1: data_size=1 (2B), tensor_dim0=tile_dim0=nelems, dim1=1.
// ---------------------------------------------------------------------------
#if __has_builtin(__builtin_amdgcn_tensor_load_to_lds) && \
    __has_builtin(__builtin_amdgcn_s_wait_tensorcnt)
#define HAVE_TDM 1
__device__ __forceinline__ void tdm_copy_elems(unsigned ldsOff, unsigned long long gaddr,
                                               unsigned nelems) {
    v4u g0;
    g0[0] = 1u;                                             // count=1 (valid)
    g0[1] = ldsOff;                                         // lds_addr
    g0[2] = (unsigned)(gaddr & 0xFFFFFFFFull);              // global_addr[31:0]
    g0[3] = (unsigned)((gaddr >> 32) & 0x01FFFFFFull) | (2u << 30);  // [56:32] | type=2
    v8i g1;
    g1[0] = (int)(1u << 16);                 // workgroup_mask=0, data_size=1 (2 bytes)
    g1[1] = (int)((nelems & 0xFFFFu) << 16); // tensor_dim0[15:0] in bits[63:48]
    g1[2] = (int)((nelems >> 16) | (1u << 16)); // tensor_dim0[31:16] | tensor_dim1=1
    g1[3] = (int)((nelems & 0xFFFFu) << 16); // tensor_dim1[31:16]=0 | tile_dim0
    g1[4] = (int)1;                          // tile_dim1=1, tile_dim2=0
    g1[5] = (int)nelems;                     // tensor_dim0_stride[31:0]
    g1[6] = 0;                               // stride0[47:32]=0, stride1[15:0]=0
    g1[7] = 0;
    v4i z4 = {0, 0, 0, 0};
#if __clang_major__ >= 23
    v8i z8 = {0, 0, 0, 0, 0, 0, 0, 0};
    __builtin_amdgcn_tensor_load_to_lds(g0, g1, z4, z4, z8, 0);
#else
    __builtin_amdgcn_tensor_load_to_lds(g0, g1, z4, z4, 0);
#endif
}
#endif

// ---------------------------------------------------------------------------
// P0a: pack fp32 [K][H] weights into bf16 panels pan[(ncat/16)*16 + ncat%16][k]
// ---------------------------------------------------------------------------
__global__ void pack_weights(const float* W0, const float* W1, const float* W2, const float* W3,
                             const float* U0, const float* U1, const float* U2, const float* U3,
                             unsigned short* Wpan, unsigned short* Upan) {
    int idx = blockIdx.x * blockDim.x + threadIdx.x;
    const int per = 4 * I * H;
    if (idx >= 2 * per) return;
    int which = idx / per;
    int rem   = idx % per;
    int g     = rem / (I * H);
    int rem2  = rem % (I * H);
    int k     = rem2 >> 9;
    int h     = rem2 & (H - 1);
    const float* Ws[4] = {W0, W1, W2, W3};
    const float* Us[4] = {U0, U1, U2, U3};
    float v = which ? Us[g][k * H + h] : Ws[g][k * H + h];
    int ncat = g * H + h;
    unsigned short* dst = which ? Upan : Wpan;
    dst[(size_t)((ncat >> 4) * 16 + (ncat & 15)) * I + k] = f2bf(v);
}

// P0b: x fp32 -> bf16 row-major, 4 elements per thread.
__global__ void cvt_x(const float* __restrict__ x, unsigned short* __restrict__ xbf) {
    int i = (blockIdx.x * blockDim.x + threadIdx.x) * 4;
    if (i >= MTOT * I) return;
    float4 v = *(const float4*)(x + i);
    unsigned lo = (unsigned)f2bf(v.x) | ((unsigned)f2bf(v.y) << 16);
    unsigned hi = (unsigned)f2bf(v.z) | ((unsigned)f2bf(v.w) << 16);
    unsigned* p = (unsigned*)(xbf + i);
    p[0] = lo; p[1] = hi;
}

// ---------------------------------------------------------------------------
// P1: gx tiles. Each block owns 4 adjacent 16-col B panels (64KB LDS), staged
// once by a single TDM descriptor; each A fragment is reused by 4 WMMAs.
// grid = 32 panel-groups x 64 mb-starts; 4 mb iterations per block.
// ---------------------------------------------------------------------------
__global__ __launch_bounds__(256) void gemm_gx(const unsigned short* __restrict__ xbf,
                                               const unsigned short* __restrict__ Wpan,
                                               const float* b0, const float* b1,
                                               const float* b2, const float* b3,
                                               unsigned short* __restrict__ gxT) {
    __shared__ __align__(16) unsigned short Bpanel[4 * 8192];   // 64 KB
    const int tid    = threadIdx.x;
    const int w      = tid >> 5;
    const int lane   = tid & 31;
    const int lanelo = lane & 15;
    const int laneHi = lane >> 4;
    const int ntg    = blockIdx.x & 31;           // panel group: nt = ntg*4 + p
    const int mb0    = blockIdx.x >> 5;           // 0..63

    const unsigned short* panelSrc = Wpan + (size_t)(ntg * 4) * 8192;
#ifdef HAVE_TDM
    if (w == 0) {
        tdm_copy_elems((unsigned)(uintptr_t)&Bpanel[0],
                       (unsigned long long)(uintptr_t)panelSrc, 4 * 8192);
        __builtin_amdgcn_s_wait_tensorcnt(0);
    }
#else
    for (int i = tid; i < (4 * 8192) / 8; i += 256)
        ((v4u*)Bpanel)[i] = ((const v4u*)panelSrc)[i];
#endif
    __syncthreads();

    const float* bs[4] = {b0, b1, b2, b3};
    float bias[4];
#pragma unroll
    for (int p = 0; p < 4; ++p) {
        int ncol = (ntg * 4 + p) * 16 + lanelo;
        bias[p] = bs[ncol >> 9][ncol & (H - 1)];
    }

    for (int mb = mb0; mb < TM / 8; mb += 64) {     // 4 macro-rows per block
        const int mt  = mb * 8 + w;
        const int row = mt * 16 + lanelo;
        const __bf16* xr = (const __bf16*)xbf + (size_t)row * I;
        v8f acc[4];
#pragma unroll
        for (int p = 0; p < 4; ++p)
            acc[p] = (v8f){0.f, 0.f, 0.f, 0.f, 0.f, 0.f, 0.f, 0.f};
        // Unroll capped at 2: keeps B fragments streaming from LDS per
        // iteration (addresses depend on k0) instead of being hoisted across
        // the mb loop into VGPRs -> no scratch spills.
#pragma unroll 2
        for (int k0 = 0; k0 < I; k0 += 32) {
            int kA = k0 + laneHi * 16;
            v8bf lo = *(const v8bf*)(xr + kA);
            v8bf hi = *(const v8bf*)(xr + kA + 8);
            v16bf a = __builtin_shufflevector(lo, hi,
                      0, 1, 2, 3, 4, 5, 6, 7, 8, 9, 10, 11, 12, 13, 14, 15);
            int kB = k0 + laneHi * 16;
#pragma unroll
            for (int p = 0; p < 4; ++p) {
                v16bf b = *(const v16bf*)((const __bf16*)Bpanel +
                          (size_t)p * 8192 + lanelo * I + kB);
                acc[p] = __builtin_amdgcn_wmma_f32_16x16x32_bf16(
                    false, a, false, b, (short)0, acc[p], false, false);
            }
        }
        // Pack C fragments as bf16 pairs; coalesced 16B store per lane per tile.
#pragma unroll
        for (int p = 0; p < 4; ++p) {
            v4u outp;
#pragma unroll
            for (int q = 0; q < 4; ++q)
                outp[q] = (unsigned)f2bf(acc[p][2 * q] + bias[p]) |
                          ((unsigned)f2bf(acc[p][2 * q + 1] + bias[p]) << 16);
            *(v4u*)(gxT + ((size_t)(mt * TN + ntg * 4 + p) * 256 + lane * 8)) = outp;
        }
    }
}

// ---------------------------------------------------------------------------
// P2: persistent recurrence. 1 workgroup, 512 threads = 16 waves.
// wave w: mi = w>>2 (batch tile), ni in [(w&3)*8, +8). c in VGPRs, h in LDS.
// ---------------------------------------------------------------------------
__global__ __launch_bounds__(512) void lstm_rec(const unsigned short* __restrict__ gxT,
                                                const unsigned short* __restrict__ Upan,
                                                float* __restrict__ hs,
                                                float* __restrict__ hfin) {
    __shared__ __align__(16) unsigned short hA[B * H];   // 64 KB bf16 h state
    const int tid    = threadIdx.x;
    const int w      = tid >> 5;
    const int lane   = tid & 31;
    const int lanelo = lane & 15;
    const int laneHi = lane >> 4;
    const int mi     = w >> 2;
    const int nbase  = (w & 3) * 8;

    for (int i = tid; i < (B * H) / 2; i += 512) ((unsigned*)hA)[i] = 0u;   // h0 = 0
    __syncthreads();

    v8f cacc[8];
#pragma unroll
    for (int j = 0; j < 8; ++j)
        cacc[j] = (v8f){0.f, 0.f, 0.f, 0.f, 0.f, 0.f, 0.f, 0.f};            // c0 = 0

    for (int t = 0; t < S; ++t) {
        unsigned hpack[8][4];
        const int mt = t * 4 + mi;
        for (int j = 0; j < 8; ++j) {
            const int ni = nbase + j;
            v8f acc4[4];
#pragma unroll
            for (int g = 0; g < 4; ++g) {
                size_t tile = (size_t)(mt * TN + (g * 32 + ni)) * 256;
                v4u p = *(const v4u*)(gxT + tile + lane * 8);
#pragma unroll
                for (int q = 0; q < 4; ++q) {
                    acc4[g][2 * q]     = bf2f(p[q] & 0xFFFFu);
                    acc4[g][2 * q + 1] = bf2f(p[q] >> 16);
                }
            }
            if (t + 1 < S)   // global_prefetch_b8: next step's i-gate tile
                __builtin_prefetch(gxT + (size_t)((mt + 4) * TN + ni) * 256 + lane * 8, 0, 0);

            const int rowA = mi * 16 + lanelo;
#pragma unroll 2
            for (int k0 = 0; k0 < H; k0 += 32) {
                int kA = k0 + laneHi * 16;
                v8bf lo = *(const v8bf*)((const __bf16*)hA + rowA * H + kA);
                v8bf hi = *(const v8bf*)((const __bf16*)hA + rowA * H + kA + 8);
                v16bf a = __builtin_shufflevector(lo, hi,
                          0, 1, 2, 3, 4, 5, 6, 7, 8, 9, 10, 11, 12, 13, 14, 15);
                int kB = k0 + laneHi * 16;
#pragma unroll
                for (int g = 0; g < 4; ++g) {
                    v16bf b = *(const v16bf*)((const __bf16*)Upan +
                              (size_t)((g * 32 + ni) * 16 + lanelo) * H + kB);
                    acc4[g] = __builtin_amdgcn_wmma_f32_16x16x32_bf16(
                        false, a, false, b, (short)0, acc4[g], false, false);
                }
            }
#pragma unroll
            for (int r = 0; r < 8; ++r) {
                float ig = sigmoidf_(acc4[0][r]);
                float fg = sigmoidf_(acc4[1][r]);
                float og = sigmoidf_(acc4[2][r]);
                float cb = tanhf(acc4[3][r]);
                float cn = fg * cacc[j][r] + ig * cb;
                cacc[j][r] = cn;
                unsigned short hb = f2bf(og * tanhf(cn));
                if (r & 1) hpack[j][r >> 1] |= ((unsigned)hb) << 16;
                else       hpack[j][r >> 1]  = (unsigned)hb;
            }
        }
        __syncthreads();   // all reads of h_{t-1} done
#pragma unroll
        for (int j = 0; j < 8; ++j) {
#pragma unroll
            for (int r = 0; r < 8; ++r) {
                int rowl = mi * 16 + laneHi * 8 + r;
                int col  = (nbase + j) * 16 + lanelo;
                unsigned short hb = (unsigned short)(hpack[j][r >> 1] >> ((r & 1) * 16));
                hA[rowl * H + col] = hb;
                float hv = bf2f(hb);
                hs[(size_t)(t * B + rowl) * H + col] = hv;
                if (t == S - 1) hfin[rowl * H + col] = hv;
            }
        }
        __syncthreads();   // h_t visible before next step
    }
}

__global__ void zero_c0(float* p, int n) {
    int i = blockIdx.x * blockDim.x + threadIdx.x;
    if (i < n) p[i] = 0.f;
}

// ---------------------------------------------------------------------------
extern "C" void kernel_launch(void* const* d_in, const int* in_sizes, int n_in,
                              void* d_out, int out_size, void* d_ws, size_t ws_size,
                              hipStream_t stream) {
    (void)in_sizes; (void)n_in; (void)out_size; (void)ws_size;

    const float* x = (const float*)d_in[0];
    const float *W0 = (const float*)d_in[1], *W1 = (const float*)d_in[2];
    const float *W2 = (const float*)d_in[3], *W3 = (const float*)d_in[4];
    const float *U0 = (const float*)d_in[5], *U1 = (const float*)d_in[6];
    const float *U2 = (const float*)d_in[7], *U3 = (const float*)d_in[8];
    const float *b0 = (const float*)d_in[9], *b1 = (const float*)d_in[10];
    const float *b2 = (const float*)d_in[11], *b3 = (const float*)d_in[12];

    uint8_t* ws = (uint8_t*)d_ws;
    unsigned short* Wpan = (unsigned short*)ws;                     // 2 MB
    unsigned short* Upan = (unsigned short*)(ws + (2ull << 20));    // 2 MB
    unsigned short* xbf  = (unsigned short*)(ws + (4ull << 20));    // 32 MB
    unsigned short* gxT  = (unsigned short*)(ws + (36ull << 20));   // 128 MB

    float* hs   = (float*)d_out;                    // [S][B][H]
    float* hfin = hs + (size_t)S * B * H;           // [B][H]
    float* c0   = hfin + (size_t)B * H;             // [B][H] zeros

    pack_weights<<<(2 * 4 * I * H + 255) / 256, 256, 0, stream>>>(
        W0, W1, W2, W3, U0, U1, U2, U3, Wpan, Upan);
    cvt_x<<<(MTOT * I / 4 + 255) / 256, 256, 0, stream>>>(x, xbf);
    gemm_gx<<<2048, 256, 0, stream>>>(xbf, Wpan, b0, b1, b2, b3, gxT);
    lstm_rec<<<1, 512, 0, stream>>>(gxT, Upan, hs, hfin);
    zero_c0<<<(B * H + 255) / 256, 256, 0, stream>>>(c0, B * H);
}
